// MolecularReactivityPredictor_34832184770798
// MI455X (gfx1250) — compile-verified, matrix operand order checked by reading, not run on hardware
//
#include <hip/hip_runtime.h>
#include <hip/hip_bf16.h>
#include <stdint.h>

typedef unsigned short u16;
typedef __attribute__((ext_vector_type(16))) __bf16 v16bf;
typedef __attribute__((ext_vector_type(8)))  float  v8f;

static const int Bn = 16384;   // batch
static const int Ed = 256;     // embedding dim
static const int Hd = 512;     // hidden dim

// ---------------- helpers ----------------
__device__ __forceinline__ u16 f2bf(float f) {
  unsigned u = __float_as_uint(f);
  u += 0x7fffu + ((u >> 16) & 1u);   // round-to-nearest-even
  return (u16)(u >> 16);
}
__device__ __forceinline__ float bf2f(u16 h) {
  return __uint_as_float(((unsigned)h) << 16);
}

union FragBF { uint4 u[2]; v16bf v; };

// generic -> LDS 32-bit offset (addrspacecast to AS3, truncate)
typedef __attribute__((address_space(3))) u16 lds_u16_t;
__device__ __forceinline__ unsigned lds_off(u16* p) {
  return (unsigned)(uintptr_t)(lds_u16_t*)p;
}

// gfx1250 async global->LDS copy, 16B per lane, tracked by ASYNCcnt
__device__ __forceinline__ void async_b128(unsigned lds, const u16* gaddr) {
  asm volatile("global_load_async_to_lds_b128 %0, %1, off"
               :: "v"(lds), "v"(gaddr) : "memory");
}
__device__ __forceinline__ void wait_async_le1() {
  asm volatile("s_wait_asynccnt 1" ::: "memory");
}
__device__ __forceinline__ void wait_async_le3() {
  asm volatile("s_wait_asynccnt 3" ::: "memory");
}
__device__ __forceinline__ void wait_async_0() {
  asm volatile("s_wait_asynccnt 0" ::: "memory");
}

// ---------------- sanitize + input LayerNorm -> bf16 ----------------
__global__ __launch_bounds__(256)
void k_prep(const float* __restrict__ x, const float* __restrict__ g,
            const float* __restrict__ b, u16* __restrict__ y, int D)
{
  int row = blockIdx.x;
  int t = threadIdx.x;
  const float* xr = x + (size_t)row * D;
  float s = 0.f, s2 = 0.f;
  for (int i = t; i < D; i += 256) {
    float v = xr[i];
    v = (isnan(v) || isinf(v)) ? 0.f : v;
    v = fminf(fmaxf(v, -10.f), 10.f);
    s += v; s2 += v * v;
  }
  __shared__ float rs[16];
  for (int o = 16; o; o >>= 1) { s += __shfl_down(s, o, 32); s2 += __shfl_down(s2, o, 32); }
  int wave = t >> 5, lane = t & 31;
  if (lane == 0) { rs[wave] = s; rs[wave + 8] = s2; }
  __syncthreads();
  if (t == 0) {
    float a = 0.f, c = 0.f;
    for (int w = 0; w < 8; w++) { a += rs[w]; c += rs[w + 8]; }
    rs[0] = a; rs[8] = c;
  }
  __syncthreads();
  float mean = rs[0] / D;
  float var  = rs[8] / D - mean * mean;
  float rstd = rsqrtf(var + 1e-5f);
  for (int i = t; i < D; i += 256) {
    float v = xr[i];
    v = (isnan(v) || isinf(v)) ? 0.f : v;
    v = fminf(fmaxf(v, -10.f), 10.f);
    y[(size_t)row * D + i] = f2bf(g[i] * (v - mean) * rstd + b[i]);
  }
}

// ---------------- LayerNorm + SiLU -> bf16 ----------------
__global__ __launch_bounds__(256)
void k_ln_silu(const float* __restrict__ x, const float* __restrict__ g,
               const float* __restrict__ b, u16* __restrict__ y, int D)
{
  int row = blockIdx.x;
  int t = threadIdx.x;
  const float* xr = x + (size_t)row * D;
  float s = 0.f, s2 = 0.f;
  for (int i = t; i < D; i += 256) { float v = xr[i]; s += v; s2 += v * v; }
  __shared__ float rs[16];
  for (int o = 16; o; o >>= 1) { s += __shfl_down(s, o, 32); s2 += __shfl_down(s2, o, 32); }
  int wave = t >> 5, lane = t & 31;
  if (lane == 0) { rs[wave] = s; rs[wave + 8] = s2; }
  __syncthreads();
  if (t == 0) {
    float a = 0.f, c = 0.f;
    for (int w = 0; w < 8; w++) { a += rs[w]; c += rs[w + 8]; }
    rs[0] = a; rs[8] = c;
  }
  __syncthreads();
  float mean = rs[0] / D;
  float var  = rs[8] / D - mean * mean;
  float rstd = rsqrtf(var + 1e-5f);
  for (int i = t; i < D; i += 256) {
    float o = g[i] * (xr[i] - mean) * rstd + b[i];
    float sig = 1.f / (1.f + __expf(-o));
    y[(size_t)row * D + i] = f2bf(o * sig);
  }
}

// -------- pack fp32 [K,N] weights -> bf16 n-major 32-k blocks --------
// Wp[((k>>5)*N + n)*32 + (k&31)] = bf16(W[k][n]); per-32-k tile is contiguous.
__global__ __launch_bounds__(256)
void k_cvt_pack(const float* __restrict__ W, u16* __restrict__ Wp, int N)
{
  __shared__ float tile[32][33];
  int n0 = blockIdx.x * 32;
  int k0 = blockIdx.y * 32;
  int t = threadIdx.x;
  #pragma unroll
  for (int e = 0; e < 4; e++) {
    int idx = t + e * 256;
    int kk = idx >> 5, nn = idx & 31;
    tile[kk][nn] = W[(size_t)(k0 + kk) * N + n0 + nn];
  }
  __syncthreads();
  #pragma unroll
  for (int e = 0; e < 4; e++) {
    int idx = t + e * 256;
    int nn = idx >> 5, kk = idx & 31;   // consecutive t -> consecutive kk (coalesced)
    Wp[((size_t)(k0 >> 5) * N + n0 + nn) * 32 + kk] = f2bf(tile[kk][nn]);
  }
}

// -------- build packed Wcat: col k = vec(W_k), col 512+k = vec(W_k^T) --------
// Row index of conceptual Wcat[65536,1024] is kglob=(i*256+j); packed as above.
__global__ __launch_bounds__(256)
void k_wcat(const float* __restrict__ W, u16* __restrict__ Wcat)
{
  __shared__ float tile[32][33];
  int i  = blockIdx.z;          // 0..255
  int k0 = blockIdx.x * 32;     // gridDim.x = 16  (512 k)
  int j0 = blockIdx.y * 32;     // gridDim.y = 8   (256 j)
  int t = threadIdx.x;
  #pragma unroll
  for (int e = 0; e < 4; e++) {
    int idx = t + e * 256;
    int kk = idx >> 5, jj = idx & 31;
    tile[kk][jj] = W[(size_t)(k0 + kk) * 65536 + (size_t)i * 256 + j0 + jj];
  }
  __syncthreads();
  // ij half: kglob = i*256 + j0+rr -> step = i*8 + j0/32, kk = rr ; n = k0+cc
  #pragma unroll
  for (int e = 0; e < 4; e++) {
    int idx = t + e * 256;
    int rr = idx & 31, cc = idx >> 5;   // coalesced over rr
    u16 v = f2bf(tile[cc][rr]);         // = W[k0+cc][i][j0+rr]
    Wcat[(((size_t)i * 8 + (j0 >> 5)) * 1024 + (k0 + cc)) * 32 + rr] = v;
  }
  // ji half: kglob = (j0+rr)*256 + i -> step = (j0+rr)*8 + (i>>5), kk = i&31 ; n = 512+k0+cc
  #pragma unroll
  for (int e = 0; e < 4; e++) {
    int idx = t + e * 256;
    int cc = idx & 31, rr = idx >> 5;
    u16 v = f2bf(tile[cc][rr]);
    Wcat[(((size_t)(j0 + rr) * 8 + (i >> 5)) * 1024 + (512 + k0 + cc)) * 32 + (i & 31)] = v;
  }
}

// ---------------- generic bf16 WMMA GEMM: C = f((acc? C:0) + A@Wp + bias) ----------------
// A: [M,K] bf16 row-major; Wp: packed (k_cvt_pack layout); C fp32; optional bf16 copy.
// flags: 1 = accumulate into C, 2 = ReLU, 4 = also write bf16 copy to Cbf
#define GBM 64
#define GBN 128
__global__ __launch_bounds__(256)
void k_gemm(const u16* __restrict__ A, const u16* __restrict__ Wp,
            const float* __restrict__ bias, float* __restrict__ C,
            u16* __restrict__ Cbf, int M, int N, int K, int flags)
{
  __shared__ __align__(16) u16 As[2][GBM * 32];   // [64][32] m-major
  __shared__ __align__(16) u16 Ws[2][GBN * 32];   // [128][32] n-major
  int m0 = blockIdx.y * GBM, n0 = blockIdx.x * GBN;
  int t = threadIdx.x, wave = t >> 5, lane = t & 31;
  int mb = wave >> 1, nbBase = (wave & 1) * 4;
  int nsteps = K >> 5;

  int r = t >> 2, pq = t & 3;
  const u16* gA = A + (size_t)(m0 + r) * K + pq * 8;
  unsigned ldsA[2] = { lds_off(&As[0][r * 32 + pq * 8]),
                       lds_off(&As[1][r * 32 + pq * 8]) };
  unsigned ldsW[2] = { lds_off(&Ws[0][t * 16]),
                       lds_off(&Ws[1][t * 16]) };

  auto issue = [&](int s, int buf) {
    async_b128(ldsA[buf], gA + (size_t)s * 32);
    const u16* w = Wp + ((size_t)s * N + n0) * 32 + t * 16;
    async_b128(ldsW[buf], w);
    async_b128(ldsW[buf] + 16, w + 8);
  };

  v8f acc[4] = {};
  issue(0, 0);
  for (int s = 0; s < nsteps; s++) {
    int cur = s & 1;
    if (s + 1 < nsteps) { issue(s + 1, cur ^ 1); wait_async_le3(); }
    else                { wait_async_0(); }
    __syncthreads();
    const u16* as = As[cur];
    const u16* ws = Ws[cur];
    FragBF fa, fb[4];
    {
      const u16* p = as + (mb * 16 + (lane & 15)) * 32 + (lane >> 4) * 8;
      fa.u[0] = *(const uint4*)p;
      fa.u[1] = *(const uint4*)(p + 16);
    }
    #pragma unroll
    for (int ti = 0; ti < 4; ti++) {
      const u16* p = ws + ((nbBase + ti) * 16 + (lane & 15)) * 32 + (lane >> 4) * 8;
      fb[ti].u[0] = *(const uint4*)p;
      fb[ti].u[1] = *(const uint4*)(p + 16);
    }
    #pragma unroll
    for (int ti = 0; ti < 4; ti++)
      acc[ti] = __builtin_amdgcn_wmma_f32_16x16x32_bf16(
          false, fa.v, false, fb[ti].v, (short)0, acc[ti], false, false);
    __syncthreads();
  }

  int mrow0 = m0 + mb * 16 + (lane >> 4) * 8;
  int nloc = lane & 15;
  #pragma unroll
  for (int ti = 0; ti < 4; ti++) {
    int ncol = n0 + (nbBase + ti) * 16 + nloc;
    float bv = bias ? bias[ncol] : 0.f;
    #pragma unroll
    for (int v = 0; v < 8; v++) {
      size_t idx = (size_t)(mrow0 + v) * N + ncol;
      float val = acc[ti][v] + bv;
      if (flags & 1) val += C[idx];
      if (flags & 2) val = fmaxf(val, 0.f);
      C[idx] = val;
      if (flags & 4) Cbf[idx] = f2bf(val);
    }
  }
}

// ---------------- bilinear as GEMM over on-the-fly outer products ----------------
// Inter[B,1024] = O @ Wcat + bias(k%512);  O[b, i*256+j] = e0[b,i]*e1[b,j]
#define BBM 128
#define BBN 128
__global__ __launch_bounds__(512)
void k_bilinear(const u16* __restrict__ e0, const u16* __restrict__ e1,
                const u16* __restrict__ Wcat, const float* __restrict__ bb,
                float* __restrict__ Inter)
{
  __shared__ __align__(16) u16 As[2][BBM * 32];
  __shared__ __align__(16) u16 Ws[2][BBN * 32];
  int m0 = blockIdx.y * BBM, n0 = blockIdx.x * BBN;
  int t = threadIdx.x, wave = t >> 5, lane = t & 31;
  int mb = wave >> 1, nbBase = (wave & 1) * 4;
  const int nsteps = 2048;                 // K = 65536 / 32

  int r = t >> 2, pq = t & 3;
  unsigned ldsW[2] = { lds_off(&Ws[0][t * 8]), lds_off(&Ws[1][t * 8]) };

  auto genA = [&](int s, int buf) {        // A tile: e0[m,i] * e1[m, j0..j0+31]
    int k0 = s * 32;
    int ii = k0 >> 8, j0 = k0 & 255;
    u16 tmp[8];
    *(uint4*)tmp = *(const uint4*)(e1 + (size_t)(m0 + r) * 256 + j0 + pq * 8);
    float e0v = bf2f(e0[(size_t)(m0 + r) * 256 + ii]);
    u16 outv[8];
    #pragma unroll
    for (int i = 0; i < 8; i++) outv[i] = f2bf(e0v * bf2f(tmp[i]));
    *(uint4*)(&As[buf][r * 32 + pq * 8]) = *(uint4*)outv;
  };
  auto issueW = [&](int s, int buf) {
    async_b128(ldsW[buf], Wcat + ((size_t)s * 1024 + n0) * 32 + t * 8);
  };

  v8f acc[4] = {};
  genA(0, 0);
  issueW(0, 0);
  for (int s = 0; s < nsteps; s++) {
    int cur = s & 1;
    if (s + 1 < nsteps) { genA(s + 1, cur ^ 1); issueW(s + 1, cur ^ 1); wait_async_le1(); }
    else                { wait_async_0(); }
    __syncthreads();
    const u16* as = As[cur];
    const u16* ws = Ws[cur];
    FragBF fa, fb[4];
    {
      const u16* p = as + (mb * 16 + (lane & 15)) * 32 + (lane >> 4) * 8;
      fa.u[0] = *(const uint4*)p;
      fa.u[1] = *(const uint4*)(p + 16);
    }
    #pragma unroll
    for (int ti = 0; ti < 4; ti++) {
      const u16* p = ws + ((nbBase + ti) * 16 + (lane & 15)) * 32 + (lane >> 4) * 8;
      fb[ti].u[0] = *(const uint4*)p;
      fb[ti].u[1] = *(const uint4*)(p + 16);
    }
    #pragma unroll
    for (int ti = 0; ti < 4; ti++)
      acc[ti] = __builtin_amdgcn_wmma_f32_16x16x32_bf16(
          false, fa.v, false, fb[ti].v, (short)0, acc[ti], false, false);
    __syncthreads();
  }

  int mrow0 = m0 + mb * 16 + (lane >> 4) * 8;
  int nloc = lane & 15;
  #pragma unroll
  for (int ti = 0; ti < 4; ti++) {
    int ncol = n0 + (nbBase + ti) * 16 + nloc;
    float bv = bb[ncol & 511];
    #pragma unroll
    for (int v = 0; v < 8; v++)
      Inter[(size_t)(mrow0 + v) * 1024 + ncol] = acc[ti][v] + bv;
  }
}

// ---------------- split Inter into ij / ji / 0.5*(ij+ji), bf16 ----------------
__global__ void k_combine(const float* __restrict__ Inter, u16* __restrict__ ij,
                          u16* __restrict__ ji, u16* __restrict__ ip)
{
  size_t idx = (size_t)blockIdx.x * 256 + threadIdx.x;
  if (idx >= (size_t)Bn * 512) return;
  size_t b = idx >> 9, k = idx & 511;
  float a = Inter[b * 1024 + k];
  float c = Inter[b * 1024 + 512 + k];
  ij[idx] = f2bf(a);
  ji[idx] = f2bf(c);
  ip[idx] = f2bf(0.5f * (a + c));
}

// ---------------- final H->1 linear: out[b] = dot(h[b,:512], w) + b0 ----------------
__global__ __launch_bounds__(256)
void k_rowdot(const u16* __restrict__ h, const float* __restrict__ w,
              const float* __restrict__ b0, float* __restrict__ out)
{
  int row = blockIdx.x * 8 + (threadIdx.x >> 5);
  int lane = threadIdx.x & 31;
  if (row >= Bn) return;
  const u16* hr = h + (size_t)row * 512;
  float s = 0.f;
  for (int i = lane; i < 512; i += 32) s += bf2f(hr[i]) * w[i];
  for (int o = 16; o; o >>= 1) s += __shfl_down(s, o, 32);
  if (lane == 0) out[row] = s + b0[0];
}

// ---------------- host ----------------
static inline void* bump(char*& p, size_t bytes) {
  void* r = (void*)p;
  p += (bytes + 255) & ~(size_t)255;
  return r;
}

extern "C" void kernel_launch(void* const* d_in, const int* in_sizes, int n_in,
                              void* d_out, int out_size, void* d_ws, size_t ws_size,
                              hipStream_t stream) {
  (void)in_sizes; (void)n_in; (void)out_size; (void)ws_size;
  const float* mol0 = (const float*)d_in[0];
  const float* mol1 = (const float*)d_in[1];
  const float* cond = (const float*)d_in[2];
  const float* bil_b   = (const float*)d_in[3];
  const float* bil_w   = (const float*)d_in[4];
  const float* ce_l1_b = (const float*)d_in[5];
  const float* ce_l1_w = (const float*)d_in[6];
  const float* ce_l2_b = (const float*)d_in[7];
  const float* ce_l2_w = (const float*)d_in[8];
  const float* ce_ln1_b = (const float*)d_in[9];
  const float* ce_ln1_g = (const float*)d_in[10];
  const float* ce_ln2_b = (const float*)d_in[11];
  const float* ce_ln2_g = (const float*)d_in[12];
  const float* ce_lnin_b = (const float*)d_in[13];
  const float* ce_lnin_g = (const float*)d_in[14];
  const float* ce_lout_b = (const float*)d_in[15];
  const float* ce_lout_w = (const float*)d_in[16];
  const float* ed_cp_b = (const float*)d_in[17];
  const float* ed_cp_w = (const float*)d_in[18];
  const float* ed_o1_b = (const float*)d_in[19];
  const float* ed_o1_w = (const float*)d_in[20];
  const float* ed_o2_b = (const float*)d_in[21];
  const float* ed_o2_w = (const float*)d_in[22];
  const float* me_l1_b = (const float*)d_in[23];
  const float* me_l1_w = (const float*)d_in[24];
  const float* me_l2_b = (const float*)d_in[25];
  const float* me_l2_w = (const float*)d_in[26];
  const float* me_ln1_b = (const float*)d_in[27];
  const float* me_ln1_g = (const float*)d_in[28];
  const float* me_ln2_b = (const float*)d_in[29];
  const float* me_ln2_g = (const float*)d_in[30];
  const float* me_lnin_b = (const float*)d_in[31];
  const float* me_lnin_g = (const float*)d_in[32];
  const float* me_lout_b = (const float*)d_in[33];
  const float* me_lout_w = (const float*)d_in[34];
  const float* pr_cp_b = (const float*)d_in[35];
  const float* pr_cp_w = (const float*)d_in[36];
  const float* pr_o1_b = (const float*)d_in[37];
  const float* pr_o1_w = (const float*)d_in[38];
  const float* pr_o2_b = (const float*)d_in[39];
  const float* pr_o2_w = (const float*)d_in[40];
  const float* pr_o3_b = (const float*)d_in[41];
  const float* pr_o3_w = (const float*)d_in[42];
  float* out = (float*)d_out;

  char* p = (char*)d_ws;
  u16* Wcat   = (u16*)bump(p, (size_t)65536 * 1024 * 2);
  u16* X0     = (u16*)bump(p, (size_t)Bn * 2048 * 2);
  u16* X1     = (u16*)bump(p, (size_t)Bn * 2048 * 2);
  u16* Cin    = (u16*)bump(p, (size_t)Bn * 128 * 2);
  u16* wME1   = (u16*)bump(p, (size_t)2048 * 512 * 2);
  u16* wME2   = (u16*)bump(p, (size_t)512 * 256 * 2);
  u16* wMEo   = (u16*)bump(p, (size_t)256 * 256 * 2);
  u16* wCE1   = (u16*)bump(p, (size_t)128 * 512 * 2);
  u16* wCE2   = (u16*)bump(p, (size_t)512 * 256 * 2);
  u16* wCEo   = (u16*)bump(p, (size_t)256 * 256 * 2);
  u16* wEDcp  = (u16*)bump(p, (size_t)256 * 512 * 2);
  u16* wEDo1  = (u16*)bump(p, (size_t)1024 * 512 * 2);
  u16* wPRcp  = (u16*)bump(p, (size_t)256 * 512 * 2);
  u16* wPRo1  = (u16*)bump(p, (size_t)1024 * 512 * 2);
  u16* wPRo2  = (u16*)bump(p, (size_t)512 * 512 * 2);
  float* H1   = (float*)bump(p, (size_t)Bn * 512 * 4);
  u16* A1     = (u16*)bump(p, (size_t)Bn * 512 * 2);
  float* H2   = (float*)bump(p, (size_t)Bn * 256 * 4);
  u16* A2     = (u16*)bump(p, (size_t)Bn * 256 * 2);
  float* Ef   = (float*)bump(p, (size_t)Bn * 256 * 4);
  u16* E0bf   = (u16*)bump(p, (size_t)Bn * 256 * 2);
  u16* E1bf   = (u16*)bump(p, (size_t)Bn * 256 * 2);
  u16* CEbf   = (u16*)bump(p, (size_t)Bn * 256 * 2);
  float* Inter = (float*)bump(p, (size_t)Bn * 1024 * 4);
  u16* IJ     = (u16*)bump(p, (size_t)Bn * 512 * 2);
  u16* JI     = (u16*)bump(p, (size_t)Bn * 512 * 2);
  u16* IP     = (u16*)bump(p, (size_t)Bn * 512 * 2);
  float* CF   = (float*)bump(p, (size_t)Bn * 512 * 4);
  u16* condE  = (u16*)bump(p, (size_t)Bn * 512 * 2);
  u16* condP  = (u16*)bump(p, (size_t)Bn * 512 * 2);
  float* T    = (float*)bump(p, (size_t)Bn * 512 * 4);
  u16* Tbf    = (u16*)bump(p, (size_t)Bn * 512 * 2);
  float* T2   = (float*)bump(p, (size_t)Bn * 512 * 4);
  u16* T2bf   = (u16*)bump(p, (size_t)Bn * 512 * 2);

  auto pack = [&](const float* src, u16* dst, int K, int N) {
    k_cvt_pack<<<dim3(N / 32, K / 32), 256, 0, stream>>>(src, dst, N);
  };
  auto gemm = [&](const u16* A, const u16* W, const float* bias, float* C,
                  u16* Cbf, int N, int K, int flags) {
    dim3 g(N / GBN, Bn / GBM);
    k_gemm<<<g, 256, 0, stream>>>(A, W, bias, C, Cbf, Bn, N, K, flags);
  };

  // weight packing (bf16, n-major 32-k blocks, async-copy friendly)
  pack(me_l1_w, wME1, 2048, 512);
  pack(me_l2_w, wME2, 512, 256);
  pack(me_lout_w, wMEo, 256, 256);
  pack(ce_l1_w, wCE1, 128, 512);
  pack(ce_l2_w, wCE2, 512, 256);
  pack(ce_lout_w, wCEo, 256, 256);
  pack(ed_cp_w, wEDcp, 256, 512);
  pack(ed_o1_w, wEDo1, 1024, 512);
  pack(pr_cp_w, wPRcp, 256, 512);
  pack(pr_o1_w, wPRo1, 1024, 512);
  pack(pr_o2_w, wPRo2, 512, 512);
  k_wcat<<<dim3(16, 8, 256), 256, 0, stream>>>(bil_w, Wcat);

  // sanitize + input LN -> bf16
  k_prep<<<Bn, 256, 0, stream>>>(mol0, me_lnin_g, me_lnin_b, X0, 2048);
  k_prep<<<Bn, 256, 0, stream>>>(mol1, me_lnin_g, me_lnin_b, X1, 2048);
  k_prep<<<Bn, 256, 0, stream>>>(cond, ce_lnin_g, ce_lnin_b, Cin, 128);

  // encoders
  auto encode = [&](const u16* X, int Din, const u16* w1, const float* b1,
                    const float* g1, const float* bb1, const u16* w2,
                    const float* b2, const float* g2, const float* bb2,
                    const u16* wo, const float* bo, u16* Ebf) {
    gemm(X, w1, b1, H1, nullptr, Hd, Din, 0);
    k_ln_silu<<<Bn, 256, 0, stream>>>(H1, g1, bb1, A1, Hd);
    gemm(A1, w2, b2, H2, nullptr, Ed, Hd, 0);
    k_ln_silu<<<Bn, 256, 0, stream>>>(H2, g2, bb2, A2, Ed);
    gemm(A2, wo, bo, Ef, Ebf, Ed, Ed, 4 /*BF16*/);
  };
  encode(X0, 2048, wME1, me_l1_b, me_ln1_g, me_ln1_b, wME2, me_l2_b,
         me_ln2_g, me_ln2_b, wMEo, me_lout_b, E0bf);
  encode(X1, 2048, wME1, me_l1_b, me_ln1_g, me_ln1_b, wME2, me_l2_b,
         me_ln2_g, me_ln2_b, wMEo, me_lout_b, E1bf);
  encode(Cin, 128, wCE1, ce_l1_b, ce_ln1_g, ce_ln1_b, wCE2, ce_l2_b,
         ce_ln2_g, ce_ln2_b, wCEo, ce_lout_b, CEbf);

  // bilinear (both orientations in one GEMM)
  k_bilinear<<<dim3(1024 / BBN, Bn / BBM), 512, 0, stream>>>(E0bf, E1bf, Wcat, bil_b, Inter);
  k_combine<<<(Bn * 512 + 255) / 256, 256, 0, stream>>>(Inter, IJ, JI, IP);

  // condition projections
  gemm(CEbf, wEDcp, ed_cp_b, CF, condE, Hd, Ed, 4);
  gemm(CEbf, wPRcp, pr_cp_b, CF, condP, Hd, Ed, 4);

  // edge head (ij)
  gemm(IJ, wEDo1, ed_o1_b, T, nullptr, Hd, Hd, 0);
  gemm(condE, wEDo1 + (size_t)512 * 512, nullptr, T, Tbf, Hd, Hd, 1 | 2 | 4);
  k_rowdot<<<Bn / 8, 256, 0, stream>>>(Tbf, ed_o2_w, ed_o2_b, out);

  // edge head (ji)
  gemm(JI, wEDo1, ed_o1_b, T, nullptr, Hd, Hd, 0);
  gemm(condE, wEDo1 + (size_t)512 * 512, nullptr, T, Tbf, Hd, Hd, 1 | 2 | 4);
  k_rowdot<<<Bn / 8, 256, 0, stream>>>(Tbf, ed_o2_w, ed_o2_b, out + Bn);

  // product head
  gemm(IP, wPRo1, pr_o1_b, T, nullptr, Hd, Hd, 0);
  gemm(condP, wPRo1 + (size_t)512 * 512, nullptr, T, Tbf, Hd, Hd, 1 | 2 | 4);
  gemm(Tbf, wPRo2, pr_o2_b, T2, T2bf, Hd, Hd, 2 | 4);
  k_rowdot<<<Bn / 8, 256, 0, stream>>>(T2bf, pr_o3_w, pr_o3_b, out + 2 * Bn);
}
